// PVLSTMComp_49615462203578
// MI455X (gfx1250) — compile-verified, matrix operand order checked by reading, not run on hardware
//
#include <hip/hip_runtime.h>
#include <hip/hip_bf16.h>

// ---------------------------------------------------------------------------
// Types for CDNA5 WMMA (gfx1250, wave32)
// ---------------------------------------------------------------------------
typedef __attribute__((ext_vector_type(16))) __bf16 v16bf;
typedef __attribute__((ext_vector_type(8)))  float  v8f;
typedef __attribute__((ext_vector_type(4)))  unsigned int v4u;

union BF16x16 { v16bf v; v4u q[2]; };

#define B_SZ   1024
#define H_SZ   1024
#define G4     4096
#define FEAT   75
#define KXP    96          // FEAT padded to multiple of 32
#define T_OBS  50
#define T_V    49
#define FUT_N  (1024*25*75)
#define CP_N   (1024*50*75)
#define CV_N   (1024*49*75)
#define CV_STR (49*75)
#define NCHUNK (KXP/32 + H_SZ/32)   // 3 + 32 = 35 K-chunks per step

__device__ __forceinline__ unsigned short f2bf(float f) {
    unsigned int u = __float_as_uint(f);
    unsigned int r = u + 0x7FFFu + ((u >> 16) & 1u);   // round-to-nearest-even
    return (unsigned short)(r >> 16);
}
__device__ __forceinline__ float fsig(float x) {
    return 1.0f / (1.0f + __expf(-x));
}

// ---------------------------------------------------------------------------
// Fused LSTM step:  gates = x@Wih^T + h@Whh^T + b ; cell update ; h out.
// Workgroup: 256 thr = 8 waves, tile = 64 batch x 64 hidden units.
// Wave (w): bblk=w&1 selects a 32-batch half (two 16-row A sub-tiles),
// cblk=w>>1 selects 16 unit cols; all 4 gate quadrants held in 8 f32
// accumulators -> each staged B tile feeds 2 WMMAs (higher reuse, halves
// L2 weight traffic vs a 32-batch tile: 128 MB/step instead of 256 MB).
// LDS staging uses GLOBAL_LOAD_ASYNC_TO_LDS_B128 (ASYNCcnt) with explicit
// double buffering: every wave issues exactly 5 async ops per K-chunk
// (1 full-EXEC x-tile op + 4 weight-tile ops), so `s_wait_asynccnt 5`
// after issuing chunk k+1 guarantees chunk k has landed (in-order completion).
// ---------------------------------------------------------------------------
__global__ __launch_bounds__(256)
void lstm_step_kernel(const unsigned short* __restrict__ x, int x_stride,
                      const unsigned short* __restrict__ h_in,     // [B][H] bf16
                      const unsigned short* __restrict__ Wih,      // [4H][KXP] bf16
                      const unsigned short* __restrict__ Whh,      // [4H][H]  bf16
                      const float* __restrict__ bih,
                      const float* __restrict__ bhh,
                      float* __restrict__ c,                       // [B][H] in/out
                      float* __restrict__ h_f32,                   // [B][H]
                      unsigned short* __restrict__ h_out)          // [B][H] bf16
{
    __shared__ __align__(16) unsigned short xs[2][64 * 32];        // 2 x 4 KB
    __shared__ __align__(16) unsigned short wsm[2][4 * 64 * 32];   // 2 x 16 KB

    const int t    = threadIdx.x;
    const int lane = t & 31;
    const int wv   = t >> 5;
    const int bblk = wv & 1;            // 32-batch half within the WG tile
    const int cblk = wv >> 1;           // 16-col block within 64 units
    const int r0   = blockIdx.x * 64;   // batch tile base
    const int u0   = blockIdx.y * 64;   // hidden-unit tile base

    // LDS byte offsets of the buffers (low 32 bits of generic pointer)
    const unsigned xsOff  = (unsigned)(unsigned long long)(&xs[0][0]);
    const unsigned wsmOff = (unsigned)(unsigned long long)(&wsm[0][0]);

    // x-tile staging: 64 rows x 4 segs = 256 pieces -> 1 full-EXEC op per wave
    const int pieceX = wv * 32 + lane;
    const int xrow = pieceX >> 2, xseg = (pieceX & 3) << 3;

    v8f a00 = {}, a01 = {};   // gate i : batch sub-tile 0 / 1
    v8f a10 = {}, a11 = {};   // gate f
    v8f a20 = {}, a21 = {};   // gate g
    v8f a30 = {}, a31 = {};   // gate o

    // per-lane WMMA operand addressing (ISA 16-bit A/B layout)
    const int arow0 = bblk * 32 + (lane & 15);        // A sub-tile 0
    const int arow1 = arow0 + 16;                     // A sub-tile 1
    const int brow  = cblk * 16 + (lane & 15);
    const int kseg  = (lane >> 4) << 3;               // 0 or 8 (halves)

    auto stage = [&](int ci, int buf) {
        const unsigned short* Xp = (ci < KXP / 32) ? x : h_in;
        const unsigned short* Wp = (ci < KXP / 32) ? Wih : Whh;
        const int xstr = (ci < KXP / 32) ? x_stride : H_SZ;
        const int wstr = (ci < KXP / 32) ? KXP : H_SZ;
        const int kb   = (ci < KXP / 32) ? ci * 32 : (ci - KXP / 32) * 32;

        {   // 1 async op per wave (all lanes active)
            unsigned lo = xsOff + (unsigned)(buf * (64 * 32) + xrow * 32 + xseg) * 2u;
            const unsigned short* ga = &Xp[(size_t)(r0 + xrow) * xstr + kb + xseg];
            asm volatile("global_load_async_to_lds_b128 %0, %1, off"
                         :: "v"(lo), "v"(ga) : "memory");
        }
        #pragma unroll
        for (int i = 0; i < 4; ++i) {          // 4 async ops per wave
            int piece = wv * 128 + i * 32 + lane;   // 0..1023
            int row = piece >> 2;                   // q*64 + j
            int seg = (piece & 3) << 3;
            int q = row >> 6, j = row & 63;
            unsigned lo = wsmOff + (unsigned)(buf * (4 * 64 * 32) + row * 32 + seg) * 2u;
            const unsigned short* ga = &Wp[(size_t)(q * H_SZ + u0 + j) * wstr + kb + seg];
            asm volatile("global_load_async_to_lds_b128 %0, %1, off"
                         :: "v"(lo), "v"(ga) : "memory");
        }
    };

    stage(0, 0);
    for (int ci = 0; ci < NCHUNK; ++ci) {
        const int buf = ci & 1;
        __syncthreads();                 // all waves done reading buf^1
        if (ci + 1 < NCHUNK) {
            stage(ci + 1, buf ^ 1);      // overlap next chunk's DMA with compute
            asm volatile("s_wait_asynccnt 0x5" ::: "memory");   // chunk ci landed
        } else {
            asm volatile("s_wait_asynccnt 0x0" ::: "memory");
        }
        __syncthreads();                 // chunk ci visible from all waves

        const unsigned short* xb = xs[buf];
        const unsigned short* wb = wsm[buf];

        BF16x16 A0, A1, Bq;
        A0.q[0] = *(const v4u*)&xb[arow0 * 32 + kseg];
        A0.q[1] = *(const v4u*)&xb[arow0 * 32 + kseg + 16];
        A1.q[0] = *(const v4u*)&xb[arow1 * 32 + kseg];
        A1.q[1] = *(const v4u*)&xb[arow1 * 32 + kseg + 16];

        Bq.q[0] = *(const v4u*)&wb[(0 * 64 + brow) * 32 + kseg];
        Bq.q[1] = *(const v4u*)&wb[(0 * 64 + brow) * 32 + kseg + 16];
        a00 = __builtin_amdgcn_wmma_f32_16x16x32_bf16(false, A0.v, false, Bq.v,
                                                      (short)0, a00, false, false);
        a01 = __builtin_amdgcn_wmma_f32_16x16x32_bf16(false, A1.v, false, Bq.v,
                                                      (short)0, a01, false, false);
        Bq.q[0] = *(const v4u*)&wb[(1 * 64 + brow) * 32 + kseg];
        Bq.q[1] = *(const v4u*)&wb[(1 * 64 + brow) * 32 + kseg + 16];
        a10 = __builtin_amdgcn_wmma_f32_16x16x32_bf16(false, A0.v, false, Bq.v,
                                                      (short)0, a10, false, false);
        a11 = __builtin_amdgcn_wmma_f32_16x16x32_bf16(false, A1.v, false, Bq.v,
                                                      (short)0, a11, false, false);
        Bq.q[0] = *(const v4u*)&wb[(2 * 64 + brow) * 32 + kseg];
        Bq.q[1] = *(const v4u*)&wb[(2 * 64 + brow) * 32 + kseg + 16];
        a20 = __builtin_amdgcn_wmma_f32_16x16x32_bf16(false, A0.v, false, Bq.v,
                                                      (short)0, a20, false, false);
        a21 = __builtin_amdgcn_wmma_f32_16x16x32_bf16(false, A1.v, false, Bq.v,
                                                      (short)0, a21, false, false);
        Bq.q[0] = *(const v4u*)&wb[(3 * 64 + brow) * 32 + kseg];
        Bq.q[1] = *(const v4u*)&wb[(3 * 64 + brow) * 32 + kseg + 16];
        a30 = __builtin_amdgcn_wmma_f32_16x16x32_bf16(false, A0.v, false, Bq.v,
                                                      (short)0, a30, false, false);
        a31 = __builtin_amdgcn_wmma_f32_16x16x32_bf16(false, A1.v, false, Bq.v,
                                                      (short)0, a31, false, false);
    }

    // epilogue: C/D layout -> lane n = lane&15 (column), rows m = r (+8 if lane>=16)
    const int n = lane & 15;
    const int u = u0 + cblk * 16 + n;

    const float bI = bih[u]            + bhh[u];
    const float bF = bih[H_SZ + u]     + bhh[H_SZ + u];
    const float bG = bih[2 * H_SZ + u] + bhh[2 * H_SZ + u];
    const float bO = bih[3 * H_SZ + u] + bhh[3 * H_SZ + u];

    const int bbase0 = r0 + bblk * 32 + ((lane >> 4) << 3);        // sub-tile 0
    #pragma unroll
    for (int r = 0; r < 8; ++r) {
        const size_t idx = (size_t)(bbase0 + r) * H_SZ + u;
        float iv = fsig(a00[r] + bI);
        float fv = fsig(a10[r] + bF);
        float gv = tanhf(a20[r] + bG);
        float ov = fsig(a30[r] + bO);
        float cn = fv * c[idx] + iv * gv;
        float hn = ov * tanhf(cn);
        c[idx]     = cn;
        h_f32[idx] = hn;
        h_out[idx] = f2bf(hn);
    }
    const int bbase1 = bbase0 + 16;                                // sub-tile 1
    #pragma unroll
    for (int r = 0; r < 8; ++r) {
        const size_t idx = (size_t)(bbase1 + r) * H_SZ + u;
        float iv = fsig(a01[r] + bI);
        float fv = fsig(a11[r] + bF);
        float gv = tanhf(a21[r] + bG);
        float ov = fsig(a31[r] + bO);
        float cn = fv * c[idx] + iv * gv;
        float hn = ov * tanhf(cn);
        c[idx]     = cn;
        h_f32[idx] = hn;
        h_out[idx] = f2bf(hn);
    }
}

// ---------------------------------------------------------------------------
// Decoder FC: out = clip(h @ fcW^T + b, -1, 1) ; also emits bf16 next-input.
// One wave per WG; 16 batch x 16 cols per wave; grid (B/16, 80/16).
// ---------------------------------------------------------------------------
__global__ __launch_bounds__(32)
void fc_kernel(const unsigned short* __restrict__ h,   // [B][H] bf16
               const unsigned short* __restrict__ W,   // [80][H] bf16 (rows 75..79 zero)
               const float* __restrict__ bias,         // [75]
               float* __restrict__ out,                // comp_vel + t*75, row stride CV_STR
               unsigned short* __restrict__ xn)        // [B][KXP] bf16
{
    const int lane  = threadIdx.x;
    const int bbase = blockIdx.x * 16;
    const int nb    = blockIdx.y * 16;
    const int kseg  = (lane >> 4) << 3;
    const int m     = lane & 15;

    const unsigned short* hrow = h + (size_t)(bbase + m) * H_SZ;
    const unsigned short* wrow = W + (size_t)(nb + m) * H_SZ;

    v8f acc = {};
    for (int kb = 0; kb < H_SZ; kb += 32) {
        BF16x16 A, Bq;
        A.q[0]  = *(const v4u*)&hrow[kb + kseg];
        A.q[1]  = *(const v4u*)&hrow[kb + kseg + 16];
        Bq.q[0] = *(const v4u*)&wrow[kb + kseg];
        Bq.q[1] = *(const v4u*)&wrow[kb + kseg + 16];
        acc = __builtin_amdgcn_wmma_f32_16x16x32_bf16(false, A.v, false, Bq.v,
                                                      (short)0, acc, false, false);
    }

    const int n = nb + m;
    if (n < FEAT) {
        const float bv = bias[n];
        const int bb = bbase + ((lane >> 4) << 3);
        #pragma unroll
        for (int r = 0; r < 8; ++r) {
            float v = acc[r] + bv;
            v = fminf(1.0f, fmaxf(-1.0f, v));
            out[(size_t)(bb + r) * CV_STR + n] = v;
            xn[(size_t)(bb + r) * KXP + n]     = f2bf(v);
        }
    }
}

// ---------------------------------------------------------------------------
// Prep / utility kernels
// ---------------------------------------------------------------------------
__global__ void zero_u32(unsigned int* p, int n) {
    int i = blockIdx.x * blockDim.x + threadIdx.x;
    if (i < n) p[i] = 0u;
}

// f32 [rows][cols] -> bf16 [rows][cpad] (zero-pad columns)
__global__ void cvt_pad(const float* __restrict__ s, unsigned short* __restrict__ d,
                        int rows, int cols, int cpad) {
    int i = blockIdx.x * blockDim.x + threadIdx.x;
    if (i >= rows * cpad) return;
    int r = i / cpad, cc = i - r * cpad;
    d[i] = (cc < cols) ? f2bf(s[(size_t)r * cols + cc]) : (unsigned short)0;
}

// f32 [srows][cols] -> bf16 [drows][cols] (zero-pad rows)
__global__ void cvt_padrows(const float* __restrict__ s, unsigned short* __restrict__ d,
                            int srows, int drows, int cols) {
    int i = blockIdx.x * blockDim.x + threadIdx.x;
    if (i >= drows * cols) return;
    int r = i / cols;
    d[i] = (r < srows) ? f2bf(s[i]) : (unsigned short)0;
}

// observed_pose f32 [B][50][75] -> bf16 [B][50][96]
__global__ void build_pose(const float* __restrict__ p, unsigned short* __restrict__ d) {
    int i = blockIdx.x * blockDim.x + threadIdx.x;
    if (i >= B_SZ * T_OBS * KXP) return;
    int f = i % KXP;
    int t = (i / KXP) % T_OBS;
    int b = i / (KXP * T_OBS);
    d[i] = (f < FEAT) ? f2bf(p[((size_t)b * T_OBS + t) * FEAT + f]) : (unsigned short)0;
}

// vel = diff(pose) with noise masking -> bf16 [B][49][96]
__global__ void build_vel(const float* __restrict__ p, const int* __restrict__ noise,
                          unsigned short* __restrict__ d) {
    int i = blockIdx.x * blockDim.x + threadIdx.x;
    if (i >= B_SZ * T_V * KXP) return;
    int f = i % KXP;
    int t = (i / KXP) % T_V;
    int b = i / (KXP * T_V);
    unsigned short o = 0;
    if (f < FEAT) {
        float v = p[((size_t)b * T_OBS + t + 1) * FEAT + f]
                - p[((size_t)b * T_OBS + t) * FEAT + f];
        int kp = f / 3;
        if (noise[((size_t)b * T_OBS + t + 1) * 25 + kp] == 1) v = 0.0f;
        o = f2bf(v);
    }
    d[i] = o;
}

// h = h_pose + h_vel ; c = c_pose + c_vel
__global__ void combine_kernel(const float* __restrict__ hp, const float* __restrict__ hv,
                               const float* __restrict__ cp, const float* __restrict__ cv,
                               float* __restrict__ hd, unsigned short* __restrict__ hdb,
                               float* __restrict__ cd) {
    int i = blockIdx.x * blockDim.x + threadIdx.x;
    if (i >= B_SZ * H_SZ) return;
    float h = hp[i] + hv[i];
    float c = cp[i] + cv[i];
    hd[i]  = h;
    hdb[i] = f2bf(h);
    cd[i]  = c;
}

// comp_pose = [pose0, pose0 + cumsum(comp_vel)]
__global__ void comp_pose_kernel(const float* __restrict__ pose,
                                 const float* __restrict__ cv,
                                 float* __restrict__ cp) {
    int i = blockIdx.x * blockDim.x + threadIdx.x;
    if (i >= B_SZ * FEAT) return;
    int b = i / FEAT, f = i - b * FEAT;
    float acc = pose[(size_t)b * T_OBS * FEAT + f];
    cp[(size_t)b * T_OBS * FEAT + f] = acc;
    for (int t = 0; t < T_V; ++t) {
        acc += cv[(size_t)b * T_V * FEAT + t * FEAT + f];
        cp[(size_t)b * T_OBS * FEAT + (t + 1) * FEAT + f] = acc;
    }
}

// ---------------------------------------------------------------------------
// Launch
// ---------------------------------------------------------------------------
extern "C" void kernel_launch(void* const* d_in, const int* in_sizes, int n_in,
                              void* d_out, int out_size, void* d_ws, size_t ws_size,
                              hipStream_t stream) {
    (void)in_sizes; (void)n_in; (void)out_size; (void)ws_size;

    const float* obs_pose  = (const float*)d_in[0];
    const float* fut_pose  = (const float*)d_in[1];
    const int*   obs_noise = (const int*)  d_in[2];
    const float* pe_Wih = (const float*)d_in[3];
    const float* pe_Whh = (const float*)d_in[4];
    const float* pe_bih = (const float*)d_in[5];
    const float* pe_bhh = (const float*)d_in[6];
    const float* ve_Wih = (const float*)d_in[7];
    const float* ve_Whh = (const float*)d_in[8];
    const float* ve_bih = (const float*)d_in[9];
    const float* ve_bhh = (const float*)d_in[10];
    const float* dc_Wih = (const float*)d_in[11];
    const float* dc_Whh = (const float*)d_in[12];
    const float* dc_bih = (const float*)d_in[13];
    const float* dc_bhh = (const float*)d_in[14];
    const float* fc_W   = (const float*)d_in[15];
    const float* fc_b   = (const float*)d_in[16];

    char* ws = (char*)d_ws;
    size_t off = 0;
    auto alloc = [&](size_t bytes) -> void* {
        off = (off + 255) & ~(size_t)255;
        void* p = ws + off;
        off += bytes;
        return p;
    };

    const size_t HB = (size_t)B_SZ * H_SZ;      // elements

    unsigned short* peWih_b = (unsigned short*)alloc((size_t)G4 * KXP * 2);
    unsigned short* veWih_b = (unsigned short*)alloc((size_t)G4 * KXP * 2);
    unsigned short* dcWih_b = (unsigned short*)alloc((size_t)G4 * KXP * 2);
    unsigned short* peWhh_b = (unsigned short*)alloc((size_t)G4 * H_SZ * 2);
    unsigned short* veWhh_b = (unsigned short*)alloc((size_t)G4 * H_SZ * 2);
    unsigned short* dcWhh_b = (unsigned short*)alloc((size_t)G4 * H_SZ * 2);
    unsigned short* fcW_b   = (unsigned short*)alloc((size_t)80 * H_SZ * 2);
    unsigned short* pose_b  = (unsigned short*)alloc((size_t)B_SZ * T_OBS * KXP * 2);
    unsigned short* vel_b   = (unsigned short*)alloc((size_t)B_SZ * T_V   * KXP * 2);
    unsigned short* hp_b    = (unsigned short*)alloc(2 * HB * 2);   // ping-pong bf16 h
    unsigned short* hv_b    = (unsigned short*)alloc(2 * HB * 2);
    unsigned short* hd_b    = (unsigned short*)alloc(2 * HB * 2);
    float* c_pose = (float*)alloc(HB * 4);
    float* c_vel  = (float*)alloc(HB * 4);
    float* c_dec  = (float*)alloc(HB * 4);
    float* h_pose = (float*)alloc(HB * 4);
    float* h_vel  = (float*)alloc(HB * 4);
    float* h_dec  = (float*)alloc(HB * 4);
    unsigned short* x_dec = (unsigned short*)alloc((size_t)B_SZ * KXP * 2);

    float* out_cp = (float*)d_out + FUT_N;           // comp_pose
    float* out_cv = (float*)d_out + FUT_N + CP_N;    // comp_vel

    const int TB = 256;
    auto g1 = [&](int n) { return dim3((n + TB - 1) / TB); };

    // --- one-time prep (per call; deterministic) ---
    cvt_pad<<<g1(G4 * KXP), TB, 0, stream>>>(pe_Wih, peWih_b, G4, FEAT, KXP);
    cvt_pad<<<g1(G4 * KXP), TB, 0, stream>>>(ve_Wih, veWih_b, G4, FEAT, KXP);
    cvt_pad<<<g1(G4 * KXP), TB, 0, stream>>>(dc_Wih, dcWih_b, G4, FEAT, KXP);
    cvt_pad<<<g1(G4 * H_SZ), TB, 0, stream>>>(pe_Whh, peWhh_b, G4, H_SZ, H_SZ);
    cvt_pad<<<g1(G4 * H_SZ), TB, 0, stream>>>(ve_Whh, veWhh_b, G4, H_SZ, H_SZ);
    cvt_pad<<<g1(G4 * H_SZ), TB, 0, stream>>>(dc_Whh, dcWhh_b, G4, H_SZ, H_SZ);
    cvt_padrows<<<g1(80 * H_SZ), TB, 0, stream>>>(fc_W, fcW_b, FEAT, 80, H_SZ);
    build_pose<<<g1(B_SZ * T_OBS * KXP), TB, 0, stream>>>(obs_pose, pose_b);
    build_vel<<<g1(B_SZ * T_V * KXP), TB, 0, stream>>>(obs_pose, obs_noise, vel_b);

    zero_u32<<<g1((int)(HB / 2)), TB, 0, stream>>>((unsigned int*)hp_b, (int)(HB / 2));
    zero_u32<<<g1((int)(HB / 2)), TB, 0, stream>>>((unsigned int*)hv_b, (int)(HB / 2));
    zero_u32<<<g1((int)HB), TB, 0, stream>>>((unsigned int*)c_pose, (int)HB);
    zero_u32<<<g1((int)HB), TB, 0, stream>>>((unsigned int*)c_vel, (int)HB);
    zero_u32<<<g1(B_SZ * KXP / 2), TB, 0, stream>>>((unsigned int*)x_dec, B_SZ * KXP / 2);

    const dim3 gs(B_SZ / 64, H_SZ / 64);   // 16 x 16 workgroups

    // --- pose encoder: 50 steps ---
    for (int t = 0; t < T_OBS; ++t) {
        lstm_step_kernel<<<gs, TB, 0, stream>>>(
            pose_b + t * KXP, T_OBS * KXP,
            hp_b + (size_t)(t & 1) * HB,
            peWih_b, peWhh_b, pe_bih, pe_bhh,
            c_pose, h_pose, hp_b + (size_t)((t + 1) & 1) * HB);
    }
    // --- velocity encoder: 49 steps ---
    for (int t = 0; t < T_V; ++t) {
        lstm_step_kernel<<<gs, TB, 0, stream>>>(
            vel_b + t * KXP, T_V * KXP,
            hv_b + (size_t)(t & 1) * HB,
            veWih_b, veWhh_b, ve_bih, ve_bhh,
            c_vel, h_vel, hv_b + (size_t)((t + 1) & 1) * HB);
    }

    // --- combine encoder states ---
    combine_kernel<<<g1((int)HB), TB, 0, stream>>>(h_pose, h_vel, c_pose, c_vel,
                                                   h_dec, hd_b, c_dec);

    // --- decoder: 49 steps, FC per step ---
    for (int t = 0; t < T_V; ++t) {
        const unsigned short* xp = (t == 0) ? (vel_b + (size_t)(T_V - 1) * KXP) : x_dec;
        const int xstr = (t == 0) ? (T_V * KXP) : KXP;
        lstm_step_kernel<<<gs, TB, 0, stream>>>(
            xp, xstr,
            hd_b + (size_t)(t & 1) * HB,
            dcWih_b, dcWhh_b, dc_bih, dc_bhh,
            c_dec, h_dec, hd_b + (size_t)((t + 1) & 1) * HB);
        fc_kernel<<<dim3(B_SZ / 16, 80 / 16), 32, 0, stream>>>(
            hd_b + (size_t)((t + 1) & 1) * HB, fcW_b, fc_b,
            out_cv + t * FEAT, x_dec);
    }

    // --- comp_pose = concat(pose0, pose0 + cumsum(comp_vel)) ---
    comp_pose_kernel<<<g1(B_SZ * FEAT), TB, 0, stream>>>(obs_pose, out_cv, out_cp);

    // --- future_pose passthrough ---
    hipMemcpyAsync(d_out, fut_pose, (size_t)FUT_N * sizeof(float),
                   hipMemcpyDeviceToDevice, stream);
}
